// Attention_26087631355937
// MI455X (gfx1250) — matrix operand order repaired and checked by reading, not compiled, not run on hardware
//
#include <hip/hip_runtime.h>
#include <hip/hip_bf16.h>

// ---- CDNA5 WMMA types ----
typedef __attribute__((ext_vector_type(16))) __bf16 bf16x16;
typedef __attribute__((ext_vector_type(8)))  __bf16 bf16x8;
typedef __attribute__((ext_vector_type(8)))  float  f32x8;
typedef __attribute__((ext_vector_type(4)))  float  f32x4;

union FragBF16 { bf16x16 v; bf16x8 h[2]; };

#define WMMA_BF16(A, Bm, C) \
  __builtin_amdgcn_wmma_f32_16x16x32_bf16(false, (A), false, (Bm), (short)0, (C), false, false)

// gfx1250 async global->LDS copy (ASYNCcnt-tracked), GV addressing.
// IMPORTANT: the LDS destination pointer is passed as an input operand so the
// shared-memory object ESCAPES — otherwise LLVM's escape analysis decides the
// "memory" clobber can't touch it, treats the LDS as never-written, folds all
// fragment loads to undef and CSEs the WMMAs away.
__device__ __forceinline__ void async_g2l_b128(const __bf16* lds_dst,
                                               unsigned int lds_off,
                                               const void* gptr) {
    asm volatile("global_load_async_to_lds_b128 %0, %1, off"
                 :: "v"(lds_off),
                    "v"((unsigned long long)(uintptr_t)gptr),
                    "v"(lds_dst)
                 : "memory");
}
__device__ __forceinline__ void wait_asynccnt0() {
    asm volatile("s_wait_asynccnt 0x0" ::: "memory");
}

// Problem constants: B=4, N=2048, D=768, H=12, Z=64, H*Z=768

// =====================================================================
// Fused masked-score + logsumexp energy (emitted FIRST for disasm view).
// S^T formulation: A = K rows, B = Q^T, so the k-reduction runs along
// the VGPR axis and each lane reduces its 8x8 elements locally.
// =====================================================================

// One 128-row K block for one wave (16 q-columns):
// 16 independent WMMAs -> 8 f32x8 accumulators, then a single
// masked max / rescale / exp-sum pass over all 64 lane-local values.
__device__ __forceinline__ void energy_block(const __bf16* __restrict__ kbase,
                                             const float* __restrict__ mrow, int k0,
                                             int ln, int hi,
                                             const FragBF16& bq0, const FragBF16& bq1,
                                             float& mrun, float& lrun)
{
    f32x8 acc[8];
    #pragma unroll
    for (int kt = 0; kt < 8; ++kt) {
        // A fragments: 16 K-rows x 64 z, split z into [0,32) and [32,64)
        const __bf16* krow = kbase + (kt * 16 + ln) * 64;
        FragBF16 a0, a1;
        a0.h[0] = *(const bf16x8*)(krow + 16 * hi);
        a0.h[1] = *(const bf16x8*)(krow + 16 * hi + 8);
        a1.h[0] = *(const bf16x8*)(krow + 32 + 16 * hi);
        a1.h[1] = *(const bf16x8*)(krow + 32 + 16 * hi + 8);

        f32x8 z = {};
        z = WMMA_BF16(a0.v, bq0.v, z);
        acc[kt] = WMMA_BF16(a1.v, bq1.v, z);
    }

    // mask multiply + block max over all 64 values held by this lane
    float bm = -3.0e38f;
    #pragma unroll
    for (int kt = 0; kt < 8; ++kt) {
        // mask[q, k] : k = k0 + kt*16 + 8*hi + i -> 8 contiguous floats
        f32x8 mv = *(const f32x8*)(mrow + k0 + kt * 16 + 8 * hi);
        acc[kt] *= mv;
        #pragma unroll
        for (int i = 0; i < 8; ++i) bm = fmaxf(bm, acc[kt][i]);
    }

    // one online-softmax rescale per 128 rows; 4 independent partial sums
    const float nm = fmaxf(mrun, bm);
    float p0 = 0.f, p1 = 0.f, p2 = 0.f, p3 = 0.f;
    #pragma unroll
    for (int kt = 0; kt < 8; kt += 4) {
        #pragma unroll
        for (int i = 0; i < 8; ++i) {
            p0 += __expf(acc[kt + 0][i] - nm);
            p1 += __expf(acc[kt + 1][i] - nm);
            p2 += __expf(acc[kt + 2][i] - nm);
            p3 += __expf(acc[kt + 3][i] - nm);
        }
    }
    lrun = lrun * __expf(mrun - nm) + ((p0 + p1) + (p2 + p3));
    mrun = nm;
}

__global__ __launch_bounds__(256)
void hop_energy_kernel(const __bf16* __restrict__ qb,
                       const __bf16* __restrict__ kb,
                       const float* __restrict__ mask,
                       float* __restrict__ out)
{
    __shared__ __align__(16) __bf16 sK[2 * 128 * 64];   // 2 x 16 KB K-tiles

    const int t    = threadIdx.x;
    const int wave = t >> 5;
    const int lane = t & 31;
    const int ln   = lane & 15;
    const int hi   = lane >> 4;

    const int qt = blockIdx.x & 15;
    const int bh = blockIdx.x >> 4;
    const int b  = bh / 12;
    const int h  = bh % 12;
    const int q0 = qt * 128;

    const int q = q0 + wave * 16 + ln;   // this lane's query column

    // B fragments (Q^T, 32x16): held in registers for the whole kernel.
    const __bf16* qrow = qb + (size_t)(b * 2048 + q) * 768 + h * 64;
    FragBF16 bq0, bq1;
    bq0.h[0] = *(const bf16x8*)(qrow + 16 * hi);
    bq0.h[1] = *(const bf16x8*)(qrow + 16 * hi + 8);
    bq1.h[0] = *(const bf16x8*)(qrow + 32 + 16 * hi);
    bq1.h[1] = *(const bf16x8*)(qrow + 32 + 16 * hi + 8);

    float mrun = -3.0e38f;   // running max (finite sentinel avoids inf-inf)
    float lrun = 0.0f;       // running sum of exp

    const float* mrow = mask + (size_t)q * 2048;

    // cooperative async-copy mapping: thread -> 64 bytes (4 x b128 per lane)
    const int crow = t >> 1;
    const int cc0  = (t & 1) * 32;
    auto issue_copy = [&](int buf, int blk) {
        const __bf16* src = kb + (size_t)(b * 2048 + blk * 128 + crow) * 768
                               + h * 64 + cc0;
        const int eoff = buf * 8192 + crow * 64 + cc0;       // element offset
        __bf16* dst = &sK[eoff];
        const unsigned int loff = (unsigned int)eoff * 2u;   // LDS byte offset
        async_g2l_b128(dst, loff +  0, src +  0);
        async_g2l_b128(dst, loff + 16, src +  8);
        async_g2l_b128(dst, loff + 32, src + 16);
        async_g2l_b128(dst, loff + 48, src + 24);
    };

    issue_copy(0, 0);
    wait_asynccnt0();
    __syncthreads();

    // blk loop unrolled by 2: LDS buffer parity is compile-time constant
    for (int blk = 0; blk < 16; blk += 2) {
        // even block: compute buf0, async-prefetch blk+1 into buf1
        issue_copy(1, blk + 1);
        energy_block(&sK[0], mrow, blk * 128, ln, hi, bq0, bq1, mrun, lrun);
        wait_asynccnt0();
        __syncthreads();

        // odd block: compute buf1, async-prefetch blk+2 into buf0
        if (blk + 2 < 16) issue_copy(0, blk + 2);
        energy_block(&sK[8192], mrow, (blk + 1) * 128, ln, hi, bq0, bq1, mrun, lrun);
        wait_asynccnt0();
        __syncthreads();
    }

    // merge the two lane-halves (same q, disjoint k sets)
    const float om = __shfl_xor(mrun, 16, 32);
    const float ol = __shfl_xor(lrun, 16, 32);
    const float nm = fmaxf(mrun, om);
    const float lt = lrun * __expf(mrun - nm) + ol * __expf(om - nm);
    const float e  = -(nm + __logf(lt));

    if (hi == 0)
        out[(size_t)(b * 12 + h) * 2048 + q] = e;
}

// =====================================================================
// Kernel 1: projection  Q = g @ Wq^T, K = g @ Wk^T  (bf16 out, [B,N,H,Z])
// 256 threads (8 waves), 128(m) x 64(j) tile for both Q and K.
// Register double-buffer: global loads for d-chunk i+1 are issued while
// the WMMAs for chunk i run.
// =====================================================================
__global__ __launch_bounds__(256)
void hop_proj_kernel(const float* __restrict__ g,
                     const float* __restrict__ wq,
                     const float* __restrict__ wk,
                     __bf16* __restrict__ qo,
                     __bf16* __restrict__ ko)
{
    __shared__ __align__(16) __bf16 sG[128 * 32];   // 8 KB
    __shared__ __align__(16) __bf16 sWq[64 * 32];   // 4 KB
    __shared__ __align__(16) __bf16 sWk[64 * 32];   // 4 KB

    const int t    = threadIdx.x;
    const int wave = t >> 5;
    const int lane = t & 31;
    const int ln   = lane & 15;
    const int hi   = lane >> 4;

    const int m0 = (blockIdx.x & 63) * 128;   // 64 m-tiles over 8192 rows
    const int j0 = (blockIdx.x >> 6) * 64;    // 12 j-tiles over 768 cols

    // cooperative-load thread mapping
    const int grow = t >> 1, gc0 = (t & 1) * 16;         // g: 16 floats/thread
    const int wrow = t >> 2, wc0 = (t & 3) * 8;          // w: 8 floats/thread

    f32x8 accq[4] = {};
    f32x8 acck[4] = {};

    f32x4 gld[4], qld[2], kld[2];
    auto load_chunk = [&](int d0) {
        const float* gs = g  + (size_t)(m0 + grow) * 768 + d0 + gc0;
        gld[0] = *(const f32x4*)(gs + 0);
        gld[1] = *(const f32x4*)(gs + 4);
        gld[2] = *(const f32x4*)(gs + 8);
        gld[3] = *(const f32x4*)(gs + 12);
        const float* qs = wq + (size_t)(j0 + wrow) * 768 + d0 + wc0;
        const float* ks = wk + (size_t)(j0 + wrow) * 768 + d0 + wc0;
        qld[0] = *(const f32x4*)(qs + 0);
        qld[1] = *(const f32x4*)(qs + 4);
        kld[0] = *(const f32x4*)(ks + 0);
        kld[1] = *(const f32x4*)(ks + 4);
    };

    load_chunk(0);

    for (int it = 0; it < 24; ++it) {
        // ---- cvt + store previously loaded chunk to LDS ----
        {
            bf16x8 p0, p1, pq, pk;
            #pragma unroll
            for (int i = 0; i < 4; ++i) {
                p0[i]     = (__bf16)gld[0][i];
                p0[4 + i] = (__bf16)gld[1][i];
                p1[i]     = (__bf16)gld[2][i];
                p1[4 + i] = (__bf16)gld[3][i];
                pq[i]     = (__bf16)qld[0][i];
                pq[4 + i] = (__bf16)qld[1][i];
                pk[i]     = (__bf16)kld[0][i];
                pk[4 + i] = (__bf16)kld[1][i];
            }
            *(bf16x8*)&sG[grow * 32 + gc0]     = p0;
            *(bf16x8*)&sG[grow * 32 + gc0 + 8] = p1;
            *(bf16x8*)&sWq[wrow * 32 + wc0]    = pq;
            *(bf16x8*)&sWk[wrow * 32 + wc0]    = pk;
        }
        __syncthreads();

        // ---- issue global loads for next chunk (hidden under WMMAs) ----
        if (it + 1 < 24) load_chunk((it + 1) * 32);

        // ---- A fragment: 16x32 bf16, rows = wave's 16 m-rows ----
        FragBF16 a;
        const __bf16* gr = &sG[(wave * 16 + ln) * 32];
        a.h[0] = *(const bf16x8*)(gr + 16 * hi);
        a.h[1] = *(const bf16x8*)(gr + 16 * hi + 8);

        #pragma unroll
        for (int tj = 0; tj < 4; ++tj) {
            FragBF16 bq, bk;
            const __bf16* wr = &sWq[(tj * 16 + ln) * 32 + 16 * hi];
            bq.h[0] = *(const bf16x8*)(wr);
            bq.h[1] = *(const bf16x8*)(wr + 8);
            const __bf16* wr2 = &sWk[(tj * 16 + ln) * 32 + 16 * hi];
            bk.h[0] = *(const bf16x8*)(wr2);
            bk.h[1] = *(const bf16x8*)(wr2 + 8);

            accq[tj] = WMMA_BF16(a.v, bq.v, accq[tj]);
            acck[tj] = WMMA_BF16(a.v, bk.v, acck[tj]);
        }
        __syncthreads();
    }

    // ---- store C tiles as bf16: VGPR i -> row (8*hi + i), lane -> col ln ----
    #pragma unroll
    for (int tj = 0; tj < 4; ++tj) {
        const int j = j0 + tj * 16 + ln;
        #pragma unroll
        for (int i = 0; i < 8; ++i) {
            const int m = m0 + wave * 16 + hi * 8 + i;
            qo[(size_t)m * 768 + j] = (__bf16)accq[tj][i];
            ko[(size_t)m * 768 + j] = (__bf16)acck[tj][i];
        }
    }
}

// =====================================================================
extern "C" void kernel_launch(void* const* d_in, const int* in_sizes, int n_in,
                              void* d_out, int out_size, void* d_ws, size_t ws_size,
                              hipStream_t stream)
{
    const float* g    = (const float*)d_in[0];   // [4,2048,768]
    const float* mask = (const float*)d_in[1];   // [2048,2048]
    const float* wq   = (const float*)d_in[2];   // [12,64,768] rows j=h*64+z
    const float* wk   = (const float*)d_in[3];   // [12,64,768]
    float* out = (float*)d_out;                  // [4,12,2048]

    __bf16* qb = (__bf16*)d_ws;                  // 8192x768 bf16 = 12.6 MB
    __bf16* kb = qb + (size_t)8192 * 768;        // another 12.6 MB

    hop_proj_kernel<<<dim3(64 * 12), 256, 0, stream>>>(g, wq, wk, qb, kb);
    hop_energy_kernel<<<dim3(4 * 12 * 16), 256, 0, stream>>>(qb, kb, mask, out);
}
